// Decoder_29180007809133
// MI455X (gfx1250) — compile-verified, matrix operand order checked by reading, not run on hardware
//
#include <hip/hip_runtime.h>

// ---------------------------------------------------------------------------
// Types for CDNA5 WMMA (wave32): bf16 A/B fragments (16 bf16 / lane), f32 acc.
// ---------------------------------------------------------------------------
typedef __attribute__((ext_vector_type(16))) __bf16 v16bf;
typedef __attribute__((ext_vector_type(8)))  __bf16 v8bf;
typedef __attribute__((ext_vector_type(8)))  float  v8f;

#define Bdim 32
#define Tdim 64
#define Hdim 1024
#define Edim 512
#define Mdim 512
#define Vdim 32000

// ---------------------------------------------------------------------------
// Load one 16-element bf16 fragment for A (16xK) or B (Kx16, stored as W rows).
// ISA 16-bit A layout: lanes 0-15: elems = K {k0+0..7, k0+16..23};
//                      lanes 16-31: elems = K {k0+8..15, k0+24..31}.
// base must point at row*ld + k0 + ((lane>>4)<<3); 16B aligned chunks.
// ---------------------------------------------------------------------------
__device__ __forceinline__ v16bf ldtile(const __bf16* base) {
  v8bf lo = *reinterpret_cast<const v8bf*>(base);
  v8bf hi = *reinterpret_cast<const v8bf*>(base + 16);
  return __builtin_shufflevector(lo, hi, 0, 1, 2, 3, 4, 5, 6, 7,
                                 8, 9, 10, 11, 12, 13, 14, 15);
}

// ---------------------------------------------------------------------------
// Generic bf16 GEMM:  out[M,N] = A[M,K] * W[N,K]^T + bias, optional tanh.
// Block = 256 threads = 8 waves; each wave owns one 16x16 tile of a 32x64
// block tile.  grid = (N/64, M/32).  Optional bf16 epilogue copies let us
// write results straight into the concatenated x / mo_in activation buffers.
// ---------------------------------------------------------------------------
__global__ __launch_bounds__(256) void wmma_gemm_kernel(
    const __bf16* __restrict__ A, int lda,
    const __bf16* __restrict__ W, int ldw,
    const float* __restrict__ bias,
    float* __restrict__ out, long long ldo,
    __bf16* __restrict__ obf0, int ldb0,
    __bf16* __restrict__ obf1, int ldb1,
    int N, int K, int act) {
  const int lane = threadIdx.x & 31;
  const int wave = threadIdx.x >> 5;
  const int m0 = blockIdx.y * 32 + (wave & 1) * 16;
  const int n0 = blockIdx.x * 64 + (wave >> 1) * 16;
  if (n0 >= N) return;

  const int kb = (lane >> 4) << 3;  // 0 or 8
  const __bf16* ap = A + (long long)(m0 + (lane & 15)) * lda + kb;
  const __bf16* bp = W + (long long)(n0 + (lane & 15)) * ldw + kb;

  v8f acc = {};
  for (int k0 = 0; k0 < K; k0 += 32) {
    v16bf a = ldtile(ap + k0);
    v16bf b = ldtile(bp + k0);
    acc = __builtin_amdgcn_wmma_f32_16x16x32_bf16(
        /*neg_a=*/false, a, /*neg_b=*/false, b,
        /*c_mod=*/(short)0, acc, /*reuse_a=*/false, /*reuse_b=*/false);
  }

  // C/D layout: VGPR r -> row m0 + r + (lane>=16 ? 8 : 0); col n0 + (lane&15)
  const int n = n0 + (lane & 15);
  const int mbase = m0 + ((lane >> 4) << 3);
  const float bv = bias ? bias[n] : 0.f;
#pragma unroll
  for (int r = 0; r < 8; ++r) {
    const int m = mbase + r;
    float v = acc[r] + bv;
    if (act == 1) v = tanhf(v);
    if (out)  out[(long long)m * ldo + n] = v;
    if (obf0) obf0[(long long)m * ldb0 + n] = (__bf16)v;
    if (obf1) obf1[(long long)m * ldb1 + n] = (__bf16)v;
  }
}

// ---------------------------------------------------------------------------
// fp32 -> bf16 conversion (weights staged once per launch; L2-resident after)
// ---------------------------------------------------------------------------
__global__ void cvt_f32_bf16(const float* __restrict__ src,
                             __bf16* __restrict__ dst, long long n) {
  long long i = (long long)blockIdx.x * blockDim.x + threadIdx.x;
  if (i < n) dst[i] = (__bf16)src[i];
}

__global__ void zero_bf16(__bf16* __restrict__ p, int n) {
  int i = blockIdx.x * blockDim.x + threadIdx.x;
  if (i < n) p[i] = (__bf16)0.f;
}

// ---------------------------------------------------------------------------
// Additive attention: e[t'] = va . tanh(a[b] + Ua_enc[b,t']), softmax over t',
// context[b] = sum_t' alpha * enc_out[b,t'].  One block per batch element.
// Writes alpha straight to d_out and context (bf16) into x / mo_in slots.
// ---------------------------------------------------------------------------
__global__ __launch_bounds__(256) void attn_kernel(
    const float* __restrict__ a,       // [32,1024]  si @ Wa^T
    const float* __restrict__ ua,      // [B*T,1024] Ua_enc
    const float* __restrict__ va,      // [1024]
    const float* __restrict__ enc,     // [B,T,2048]
    float* __restrict__ alpha_out,     // alphas base [B,T,T]
    __bf16* __restrict__ x_ctx,        // x_bf  + 512   (ld 2560)
    __bf16* __restrict__ mo_ctx,       // mo_bf + 1024  (ld 3584)
    int t_step) {
  __shared__ float sh_e[Tdim];
  __shared__ float sh_alpha[Tdim];
  const int b = blockIdx.x;
  const int lane = threadIdx.x & 31;
  const int wave = threadIdx.x >> 5;
  const float* ab = a + b * Hdim;

#pragma unroll
  for (int i = 0; i < 8; ++i) {
    const int tp = wave * 8 + i;
    const float* u = ua + (long long)(b * Tdim + tp) * Hdim;
    float s = 0.f;
    for (int h = lane; h < Hdim; h += 32) s += va[h] * tanhf(ab[h] + u[h]);
    for (int off = 16; off > 0; off >>= 1) s += __shfl_xor(s, off, 32);
    if (lane == 0) sh_e[tp] = s;
  }
  __syncthreads();

  if (threadIdx.x == 0) {
    float mx = sh_e[0];
    for (int i = 1; i < Tdim; ++i) mx = fmaxf(mx, sh_e[i]);
    float sum = 0.f;
    for (int i = 0; i < Tdim; ++i) {
      float ex = expf(sh_e[i] - mx);
      sh_alpha[i] = ex;
      sum += ex;
    }
    const float inv = 1.f / sum;
    for (int i = 0; i < Tdim; ++i) sh_alpha[i] *= inv;
  }
  __syncthreads();

  if (threadIdx.x < Tdim)
    alpha_out[((long long)b * Tdim + t_step) * Tdim + threadIdx.x] =
        sh_alpha[threadIdx.x];

  const float* eb = enc + (long long)b * Tdim * (2 * Hdim);
  for (int f = threadIdx.x; f < 2 * Hdim; f += 256) {
    float c = 0.f;
    for (int tp = 0; tp < Tdim; ++tp) c += sh_alpha[tp] * eb[tp * 2 * Hdim + f];
    const __bf16 cb = (__bf16)c;
    x_ctx[b * 2560 + f] = cb;
    mo_ctx[b * 3584 + f] = cb;
  }
}

// ---------------------------------------------------------------------------
// GRU cell elementwise: r,z,n gates + state update; fans out bf16 state.
// ---------------------------------------------------------------------------
__global__ void gru_elem(const float* __restrict__ gx,
                         const float* __restrict__ gh,
                         float* __restrict__ si,
                         __bf16* __restrict__ state_bf,
                         __bf16* __restrict__ ybuf_bf,
                         __bf16* __restrict__ mo_si) {
  const int idx = blockIdx.x * blockDim.x + threadIdx.x;
  if (idx >= Bdim * Hdim) return;
  const int b = idx >> 10, h = idx & (Hdim - 1);
  const float* gxb = gx + b * 3 * Hdim;
  const float* ghb = gh + b * 3 * Hdim;
  const float r = 1.f / (1.f + expf(-(gxb[h] + ghb[h])));
  const float z = 1.f / (1.f + expf(-(gxb[Hdim + h] + ghb[Hdim + h])));
  const float n = tanhf(gxb[2 * Hdim + h] + r * ghb[2 * Hdim + h]);
  const float s = (1.f - z) * n + z * si[idx];
  si[idx] = s;
  const __bf16 sb = (__bf16)s;
  state_bf[idx] = sb;
  ybuf_bf[idx] = sb;
  mo_si[b * 3584 + h] = sb;
}

// maxout pool=2 over adjacent columns
__global__ void maxout_elem(const float* __restrict__ mo,
                            __bf16* __restrict__ m_bf) {
  const int idx = blockIdx.x * blockDim.x + threadIdx.x;
  if (idx >= Bdim * Mdim) return;
  const int b = idx >> 9, j = idx & (Mdim - 1);
  const float v = fmaxf(mo[b * 2 * Mdim + 2 * j], mo[b * 2 * Mdim + 2 * j + 1]);
  m_bf[idx] = (__bf16)v;
}

// ---------------------------------------------------------------------------
extern "C" void kernel_launch(void* const* d_in, const int* in_sizes, int n_in,
                              void* d_out, int out_size, void* d_ws,
                              size_t ws_size, hipStream_t stream) {
  const float* enc_out  = (const float*)d_in[0];
  const float* hid_enc  = (const float*)d_in[1];
  const float* Ws_w     = (const float*)d_in[2];
  const float* Ws_b     = (const float*)d_in[3];
  const float* emb_w    = (const float*)d_in[4];
  const float* emb_b    = (const float*)d_in[5];
  const float* gru_w_ih = (const float*)d_in[6];
  const float* gru_w_hh = (const float*)d_in[7];
  const float* gru_b_ih = (const float*)d_in[8];
  const float* gru_b_hh = (const float*)d_in[9];
  const float* attn_Wa  = (const float*)d_in[10];
  const float* attn_Ua  = (const float*)d_in[11];
  const float* attn_va  = (const float*)d_in[12];
  const float* mo_w     = (const float*)d_in[13];
  const float* mo_b     = (const float*)d_in[14];
  const float* fc_w     = (const float*)d_in[15];
  const float* fc_b     = (const float*)d_in[16];
  (void)in_sizes; (void)n_in; (void)out_size; (void)ws_size;

  // ---- workspace bump allocator (256B aligned) ----
  char* ws = (char*)d_ws;
  size_t off = 0;
  auto alloc = [&](size_t bytes) -> void* {
    void* p = ws + off;
    off = (off + bytes + 255) & ~(size_t)255;
    return p;
  };
  auto abf = [&](long long n) { return (__bf16*)alloc((size_t)n * 2); };
  auto af32 = [&](long long n) { return (float*)alloc((size_t)n * 4); };

  __bf16* bWs   = abf(1024LL * 1024);
  __bf16* bEmb  = abf(512LL * 1024);
  __bf16* bGih  = abf(3072LL * 2560);
  __bf16* bGhh  = abf(3072LL * 1024);
  __bf16* bWa   = abf(1024LL * 1024);
  __bf16* bUa   = abf(1024LL * 2048);
  __bf16* bMo   = abf(1024LL * 3584);
  __bf16* bFc   = abf((long long)Vdim * Mdim);
  __bf16* bEnc  = abf((long long)Bdim * Tdim * 2 * Hdim);
  __bf16* bHid  = abf(Bdim * Hdim);
  __bf16* state = abf(Bdim * Hdim);        // si (bf16)
  __bf16* ybuf  = abf(Bdim * Hdim);        // yi (bf16)
  __bf16* x_bf  = abf(Bdim * 2560);        // [yi_emb | ctx]
  __bf16* moin  = abf(Bdim * 3584);        // [si | ctx | yi_emb]
  __bf16* m_bf  = abf(Bdim * Mdim);
  float* UaEnc  = af32((long long)Bdim * Tdim * Hdim);
  float* si_f   = af32(Bdim * Hdim);
  float* a_f    = af32(Bdim * Hdim);
  float* gx     = af32(Bdim * 3 * Hdim);
  float* gh     = af32(Bdim * 3 * Hdim);
  float* mo_f   = af32(Bdim * 2 * Mdim);

  float* logits = (float*)d_out;                               // [B,T,V]
  float* alphas = logits + (long long)Bdim * Tdim * Vdim;      // [B,T,T]

  auto cvt = [&](const float* s, __bf16* d, long long n) {
    cvt_f32_bf16<<<(int)((n + 255) / 256), 256, 0, stream>>>(s, d, n);
  };
  auto gemm = [&](const __bf16* A, int lda, const __bf16* W, int ldw,
                  const float* bias, float* out, long long ldo, __bf16* o0,
                  int l0, __bf16* o1, int l1, int M, int N, int K, int act) {
    dim3 g(N / 64, M / 32);
    wmma_gemm_kernel<<<g, 256, 0, stream>>>(A, lda, W, ldw, bias, out, ldo, o0,
                                            l0, o1, l1, N, K, act);
  };

  // ---- stage bf16 weights / inputs (then L2-resident for all 64 steps) ----
  cvt(Ws_w, bWs, 1024LL * 1024);
  cvt(emb_w, bEmb, 512LL * 1024);
  cvt(gru_w_ih, bGih, 3072LL * 2560);
  cvt(gru_w_hh, bGhh, 3072LL * 1024);
  cvt(attn_Wa, bWa, 1024LL * 1024);
  cvt(attn_Ua, bUa, 1024LL * 2048);
  cvt(mo_w, bMo, 1024LL * 3584);
  cvt(fc_w, bFc, (long long)Vdim * Mdim);
  cvt(enc_out, bEnc, (long long)Bdim * Tdim * 2 * Hdim);
  cvt(hid_enc + (long long)Bdim * Hdim, bHid, Bdim * Hdim);  // hidden_enc[1]
  zero_bf16<<<(Bdim * Hdim + 255) / 256, 256, 0, stream>>>(ybuf, Bdim * Hdim);

  // si0 = tanh(hidden_enc[1] @ Ws^T + b)
  gemm(bHid, Hdim, bWs, Hdim, Ws_b, si_f, Hdim, state, Hdim, nullptr, 0, Bdim,
       Hdim, Hdim, /*tanh*/ 1);
  // Ua_enc = enc_out @ Ua^T   (2048 x 2048 x 1024)
  gemm(bEnc, 2 * Hdim, bUa, 2 * Hdim, nullptr, UaEnc, Hdim, nullptr, 0,
       nullptr, 0, Bdim * Tdim, Hdim, 2 * Hdim, 0);

  for (int t = 0; t < Tdim; ++t) {
    // a = si @ Wa^T
    gemm(state, Hdim, bWa, Hdim, nullptr, a_f, Hdim, nullptr, 0, nullptr, 0,
         Bdim, Hdim, Hdim, 0);
    // attention: e, softmax, context (+ alpha to d_out, ctx to x / mo_in)
    attn_kernel<<<Bdim, 256, 0, stream>>>(a_f, UaEnc, attn_va, enc_out, alphas,
                                          x_bf + Edim, moin + Hdim, t);
    // yi_emb = yi @ emb^T + b  -> x[:, :512] and mo_in[:, 3072:]
    gemm(ybuf, Hdim, bEmb, Hdim, emb_b, nullptr, 0, x_bf, 2560, moin + 3072,
         3584, Bdim, Edim, Hdim, 0);
    // gx = x @ W_ih^T + b_ih ; gh = si @ W_hh^T + b_hh
    gemm(x_bf, 2560, bGih, 2560, gru_b_ih, gx, 3 * Hdim, nullptr, 0, nullptr,
         0, Bdim, 3 * Hdim, 2560, 0);
    gemm(state, Hdim, bGhh, Hdim, gru_b_hh, gh, 3 * Hdim, nullptr, 0, nullptr,
         0, Bdim, 3 * Hdim, Hdim, 0);
    gru_elem<<<(Bdim * Hdim + 255) / 256, 256, 0, stream>>>(gx, gh, si_f,
                                                            state, ybuf, moin);
    // maxout input GEMM then pool
    gemm(moin, 3584, bMo, 3584, mo_b, mo_f, 2 * Mdim, nullptr, 0, nullptr, 0,
         Bdim, 2 * Mdim, 3584, 0);
    maxout_elem<<<(Bdim * Mdim + 255) / 256, 256, 0, stream>>>(mo_f, m_bf);
    // logits[:, t, :] = m @ fc^T + b
    gemm(m_bf, Mdim, bFc, Mdim, fc_b, logits + (long long)t * Vdim,
         (long long)Tdim * Vdim, nullptr, 0, nullptr, 0, Bdim, Vdim, Mdim, 0);
  }
}
